// LAS_48928267436220
// MI455X (gfx1250) — matrix operand order repaired and checked by reading, not compiled, not run on hardware
//
#include <hip/hip_runtime.h>
#include <hip/hip_bf16.h>

// ---------------------------------------------------------------------------
// LAS attention decoder for MI455X (gfx1250, wave32, WMMA + TDM).
//   B=64, T=1024, D=256, U-1=99, VOCAB=4233 (padded to 4352 for tiling).
// - All GEMMs: v_wmma_f32_16x16x32_bf16 (bf16 in / f32 accumulate).
// - att_h and eout cached as bf16 (32MB each): the 99-step scan streams from
//   the 192MB L2 instead of HBM.
// - Weights pre-packed in WMMA-fragment order: B-fragment load = 2x b128.
// - Classifier GEMM + decoder recurrence use the Tensor Data Mover
//   (tensor_load_to_lds, double-buffered LDS A tiles, s_wait_tensorcnt
//   pipelining) when the builtin is available.
// Workspace use: ~91 MB.
// ---------------------------------------------------------------------------

typedef __attribute__((ext_vector_type(16))) __bf16        v16bf;
typedef __attribute__((ext_vector_type(8)))  float         v8f;
typedef __attribute__((ext_vector_type(4)))  unsigned int  u32x4;
typedef __attribute__((ext_vector_type(4)))  int           i32x4;
typedef __attribute__((ext_vector_type(8)))  int           i32x8;

#if defined(__has_builtin)
#  if __has_builtin(__builtin_amdgcn_tensor_load_to_lds)
#    define HAVE_TDM 1
#  endif
#endif
#ifndef HAVE_TDM
#  define HAVE_TDM 0
#endif

#define DEV static __device__ __forceinline__

DEV unsigned short f2bf(float f) {
  unsigned int u = __builtin_bit_cast(unsigned int, f);
  u += 0x7fffu + ((u >> 16) & 1u);           // round-to-nearest-even
  return (unsigned short)(u >> 16);
}
DEV unsigned int f2bf_pk(float lo, float hi) {
  return (unsigned int)f2bf(lo) | ((unsigned int)f2bf(hi) << 16);
}
DEV float bf2f(unsigned int h) {
  return __builtin_bit_cast(float, h << 16);
}
DEV float sigm(float x) { return 1.0f / (1.0f + __expf(-x)); }

DEV float wsum(float v) {
#pragma unroll
  for (int o = 16; o > 0; o >>= 1) v += __shfl_down(v, o, 32);
  return v;
}

union Frag {                       // all-POD: safe in a union
  unsigned short h[16];
  u32x4          q[2];
  v16bf          v;
};

// A fragment, 16x32 bf16, source row-major [m][k], leading dim ld (halfs).
// ISA: lanes 0-15 m=0..15 / K 0..7 & 16..23 ; lanes 16-31 K 8..15 & 24..31.
DEV v16bf frag_a_rowmaj(const unsigned short* A, int ld, int m0, int k0, int lane) {
  int m   = m0 + (lane & 15);
  int khi = (lane & 16) ? 8 : 0;
  Frag f;
  f.q[0] = *(const u32x4*)(A + m * ld + k0 + khi);
  f.q[1] = *(const u32x4*)(A + m * ld + k0 + 16 + khi);
  return f.v;
}

// B fragment from weights pre-packed in fragment order: block (kt,nt) holds
// 512 halfs; lane L's 16 halfs contiguous -> 2x coalesced b128 loads.
DEV v16bf frag_b_pack(const unsigned short* W, int NT, int kt, int nt, int lane) {
  const unsigned short* p = W + (((size_t)(kt * NT + nt) * 32 + lane) << 4);
  Frag f;
  f.q[0] = *(const u32x4*)p;
  f.q[1] = *(const u32x4*)(p + 8);
  return f.v;
}

DEV v8f wmma(v16bf a, v16bf b, v8f c) {
  return __builtin_amdgcn_wmma_f32_16x16x32_bf16(false, a, false, b,
                                                 (short)0, c, false, false);
}

#if HAVE_TDM
// TDM load of a 2-D tile (tile_d0 elems/row x tile_d1 rows, 2-byte elements,
// row stride `stride` elems) from global into densely packed LDS at lds_off.
DEV void tdm_load_2d(unsigned lds_off, const unsigned short* gsrc,
                     unsigned tile_d0, unsigned tile_d1,
                     unsigned long long stride) {
  unsigned long long ga = (unsigned long long)gsrc;
  unsigned td0 = (unsigned)stride;       // tensor_dim0 = full row length
  u32x4 g0;
  g0[0] = 1u;                                            // count=1 (valid D#)
  g0[1] = lds_off;                                       // lds_addr
  g0[2] = (unsigned)(ga & 0xffffffffu);                  // global_addr[31:0]
  g0[3] = (unsigned)((ga >> 32) & 0x01ffffffu) | (2u << 30);  // addr[56:32], type=2
  i32x8 g1;
  g1[0] = 0x00010000;                                    // data_size=1 (2 bytes)
  g1[1] = (int)((td0 & 0xffffu) << 16);                  // tensor_dim0 lo16
  g1[2] = (int)((td0 >> 16) & 0xffffu);                  // dim0 hi16 | dim1 lo16(=0)
  g1[3] = (int)(0x10u | (tile_d0 << 16));                // dim1 hi16 (dim1=1M), tile_dim0
  g1[4] = (int)(tile_d1 & 0xffffu);                      // tile_dim1 (tile_dim2=0)
  g1[5] = (int)(stride & 0xffffffffull);                 // tensor_dim0_stride lo32
  g1[6] = (int)((stride >> 32) & 0xffffull);             // stride hi16
  g1[7] = 0;
  i32x4 z4 = {0, 0, 0, 0};
#if __clang_major__ >= 23
  i32x8 z8 = {0, 0, 0, 0, 0, 0, 0, 0};
  __builtin_amdgcn_tensor_load_to_lds(g0, g1, z4, z4, z8, 0);
#else
  __builtin_amdgcn_tensor_load_to_lds(g0, g1, z4, z4, 0);
#endif
}
DEV unsigned lds_off_of(const void* p) {
  return (unsigned)(unsigned long long)p;
}
#endif

// ---------------------------------------------------------------------------
// prep: pack B matrix (in[n*K+k], n<N rows, K cols) into WMMA fragment order.
// ---------------------------------------------------------------------------
__global__ __launch_bounds__(256) void k_pack(const float* in, unsigned short* outp,
                                              int N, int K, int NT) {
  int idx = blockIdx.x * 256 + threadIdx.x;
  if (idx >= (K >> 5) * NT * 512) return;
  int e = idx & 15, lane = (idx >> 4) & 31, blk = idx >> 9;
  int kt = blk / NT, nt = blk - kt * NT;
  int k = kt * 32 + ((lane & 16) ? 8 : 0) + (e & 7) + ((e & 8) ? 16 : 0);
  int n = nt * 16 + (lane & 15);
  outp[idx] = (n < N) ? f2bf(in[(size_t)n * K + k]) : (unsigned short)0;
}

// f32 -> bf16 bulk convert, 8 elements/thread, b128 in / b128 out
__global__ __launch_bounds__(256) void k_cvt(const float* in, unsigned short* outp,
                                             int n8) {
  int i = blockIdx.x * 256 + threadIdx.x;
  if (i >= n8) return;
  const float* s = in + (size_t)i * 8;
  u32x4 pk;
#pragma unroll
  for (int j = 0; j < 4; ++j) pk[j] = f2bf_pk(s[2 * j], s[2 * j + 1]);
  *(u32x4*)(outp + (size_t)i * 8) = pk;
}

__global__ __launch_bounds__(256) void k_zero(float* p, int n) {
  int i = blockIdx.x * 256 + threadIdx.x;
  if (i < n) p[i] = 0.f;
}

// embed_b[b,u,d] = bf16(emb[y[b,u], d]), u < 99 (packed 2/dword)
__global__ __launch_bounds__(256) void k_embed(const int* y, const float* emb,
                                               unsigned short* eb) {
  int idx = blockIdx.x * 256 + threadIdx.x;       // over 64*99*128 dwords
  if (idx >= 64 * 99 * 128) return;
  int d2 = idx & 127, bu = idx >> 7;
  int u = bu % 99, b = bu / 99;
  const float* src = emb + (size_t)y[b * 100 + u] * 256 + d2 * 2;
  ((unsigned int*)eb)[idx] = f2bf_pk(src[0], src[1]);
}

// ---------------------------------------------------------------------------
// att_h = eout @ w_w^T + w_b  (M=65536, N=256, K=256), stored bf16.
// ---------------------------------------------------------------------------
__global__ __launch_bounds__(256) void k_atth(const float* eout, const unsigned short* wwP,
                                              const float* w_b, unsigned short* att_h) {
  __shared__ __align__(16) unsigned short lA[128 * 32];
  const int m0 = blockIdx.x * 128, n0 = blockIdx.y * 64;
  const int tid = threadIdx.x, lane = tid & 31, wv = tid >> 5;
  const int wm = wv & 3, wn = wv >> 2;                 // 4x2 wave grid, 32x32 each
  const v8f z = {0, 0, 0, 0, 0, 0, 0, 0};
  v8f acc[2][2] = {{z, z}, {z, z}};

  for (int kt = 0; kt < 8; ++kt) {
    int k0 = kt * 32;
    {   // stage A (128x32): b128 f32 loads, packed bf16 b128 stores
      int row = tid >> 1, cg = (tid & 1) * 16;
      const float* src = eout + (size_t)(m0 + row) * 256 + k0 + cg;
      if (kt < 7) __builtin_prefetch(src + 32, 0, 1);
      u32x4 p0, p1;
#pragma unroll
      for (int j = 0; j < 4; ++j) p0[j] = f2bf_pk(src[2 * j], src[2 * j + 1]);
#pragma unroll
      for (int j = 0; j < 4; ++j) p1[j] = f2bf_pk(src[8 + 2 * j], src[8 + 2 * j + 1]);
      u32x4* dst = (u32x4*)(lA + row * 32 + cg);
      dst[0] = p0;
      dst[1] = p1;
    }
    __syncthreads();
    v16bf a0 = frag_a_rowmaj(lA, 32, wm * 32, 0, lane);
    v16bf a1 = frag_a_rowmaj(lA, 32, wm * 32 + 16, 0, lane);
    int ntb = blockIdx.y * 4 + wn * 2;
    v16bf b0 = frag_b_pack(wwP, 16, kt, ntb, lane);
    v16bf b1 = frag_b_pack(wwP, 16, kt, ntb + 1, lane);
    acc[0][0] = wmma(a0, b0, acc[0][0]);
    acc[0][1] = wmma(a0, b1, acc[0][1]);
    acc[1][0] = wmma(a1, b0, acc[1][0]);
    acc[1][1] = wmma(a1, b1, acc[1][1]);
    __syncthreads();
  }
#pragma unroll
  for (int fm = 0; fm < 2; ++fm)
#pragma unroll
    for (int fn = 0; fn < 2; ++fn) {
      int nn = n0 + wn * 32 + fn * 16 + (lane & 15);
      float bias = w_b[nn];
      int mb = m0 + wm * 32 + fm * 16 + ((lane & 16) ? 8 : 0);
#pragma unroll
      for (int r = 0; r < 8; ++r)
        att_h[(size_t)(mb + r) * 256 + nn] = f2bf(acc[fm][fn][r] + bias);
    }
}

// ---------------------------------------------------------------------------
// Attention LSTM cell + state projection for one step u.
// grid = 4 WGs, each 16 batch rows.  Gate GEMM: M=16, N=1024, K=512+256.
// ---------------------------------------------------------------------------
__global__ __launch_bounds__(256) void k_att_gates(
    int u, const unsigned short* embed_b, const unsigned short* WihP,
    const unsigned short* WhhP, const unsigned short* vwP,
    const float* bih, const float* bhh, const float* v_b, const float* y_mask,
    float* hbuf, float* cbuf, const float* ctxbuf, float* statebuf,
    unsigned short* dout_b) {
  __shared__ __align__(16) unsigned char smem[65536];
  unsigned short* xA = (unsigned short*)smem;             // 16x512 bf16
  unsigned short* hA = (unsigned short*)(smem + 16384);   // 16x256 bf16
  float* gts = (float*)smem;                              // 16x1024 f32 (reuse)
  const int b0 = blockIdx.x * 16;
  const int tid = threadIdx.x, lane = tid & 31, wv = tid >> 5;
  const v8f z = {0, 0, 0, 0, 0, 0, 0, 0};

  // phase 0: x = [embed_u | ctx], h  (bf16 tiles)
  for (int idx = tid; idx < 16 * 256; idx += 256) {
    int m = idx >> 8, k = idx & 255, b = b0 + m;
    xA[m * 512 + k]       = embed_b[(size_t)(b * 99 + u) * 256 + k];
    xA[m * 512 + 256 + k] = f2bf(ctxbuf[b * 256 + k]);
    hA[m * 256 + k]       = f2bf(hbuf[b * 256 + k]);
  }
  __syncthreads();

  // phase 1: gates = x @ WihT + h @ WhhT  (wave wv owns cols [wv*128, +128))
  v8f acc[8];
#pragma unroll
  for (int nf = 0; nf < 8; ++nf) acc[nf] = z;
  for (int kt = 0; kt < 16; ++kt) {
    v16bf a = frag_a_rowmaj(xA, 512, 0, kt * 32, lane);
#pragma unroll
    for (int nf = 0; nf < 8; ++nf)
      acc[nf] = wmma(a, frag_b_pack(WihP, 64, kt, wv * 8 + nf, lane), acc[nf]);
  }
  for (int kt = 0; kt < 8; ++kt) {
    v16bf a = frag_a_rowmaj(hA, 256, 0, kt * 32, lane);
#pragma unroll
    for (int nf = 0; nf < 8; ++nf)
      acc[nf] = wmma(a, frag_b_pack(WhhP, 64, kt, wv * 8 + nf, lane), acc[nf]);
  }
  __syncthreads();                       // xA/hA dead -> reuse as gts

  // phase 2: dump gates + biases
  const int n0w = wv * 128;
#pragma unroll
  for (int nf = 0; nf < 8; ++nf) {
    int n = n0w + nf * 16 + (lane & 15);
    float bias = bih[n] + bhh[n];
    int mrow = (lane & 16) ? 8 : 0;
#pragma unroll
    for (int r = 0; r < 8; ++r) gts[(mrow + r) * 1024 + n] = acc[nf][r] + bias;
  }
  __syncthreads();

  // phase 3: LSTM elementwise; thread tid owns column d=tid for all 16 rows
  float hn[16];
  {
    int d = tid;
#pragma unroll
    for (int m = 0; m < 16; ++m) {
      int b = b0 + m;
      float gi = gts[m * 1024 + d];
      float gf = gts[m * 1024 + 256 + d];
      float gg = gts[m * 1024 + 512 + d];
      float go = gts[m * 1024 + 768 + d];
      float c = sigm(gf) * cbuf[b * 256 + d] + sigm(gi) * tanhf(gg);
      float h = sigm(go) * tanhf(c);
      cbuf[b * 256 + d] = c;
      hbuf[b * 256 + d] = h;
      hn[m] = h;
      float ym = y_mask[b * 100 + u + 1];
      dout_b[(size_t)(b * 99 + u) * 768 + d] = f2bf(h * ym);   // lstms * ym
    }
  }
  __syncthreads();

  // phase 4: state = h_new @ vwT + v_b  (M=16, N=256, K=256)
  unsigned short* hA2 = (unsigned short*)smem;   // 16x256 bf16
#pragma unroll
  for (int m = 0; m < 16; ++m) hA2[m * 256 + tid] = f2bf(hn[m]);
  __syncthreads();
  v8f acc2[2] = {z, z};
  for (int kt = 0; kt < 8; ++kt) {
    v16bf a = frag_a_rowmaj(hA2, 256, 0, kt * 32, lane);
#pragma unroll
    for (int nf = 0; nf < 2; ++nf)
      acc2[nf] = wmma(a, frag_b_pack(vwP, 16, kt, wv * 2 + nf, lane), acc2[nf]);
  }
#pragma unroll
  for (int nf = 0; nf < 2; ++nf) {
    int n = wv * 32 + nf * 16 + (lane & 15);
    float bias = v_b[n];
    int mrow = (lane & 16) ? 8 : 0;
#pragma unroll
    for (int r = 0; r < 8; ++r)
      statebuf[(b0 + mrow + r) * 256 + n] = acc2[nf][r] + bias;
  }
}

// ---------------------------------------------------------------------------
// Attention scores + softmax + context for one step u.  grid = 64 WGs.
// ---------------------------------------------------------------------------
__global__ __launch_bounds__(256) void k_att_score(
    int u, const unsigned short* att_h, const unsigned short* eout_b,
    const float* statebuf, const float* w_att_v, const float* x_mask,
    const float* y_mask, float* ctxbuf, unsigned short* dout_b) {
  __shared__ float sc[1024];
  __shared__ float red[8];
  const int b = blockIdx.x;
  const int tid = threadIdx.x, lane = tid & 31, wv = tid >> 5;

  float st[8], wvv[8];
#pragma unroll
  for (int i = 0; i < 8; ++i) {
    int dd = lane * 8 + i;                 // lane covers 8 contiguous d
    st[i]  = statebuf[b * 256 + dd];
    wvv[i] = w_att_v[dd];
  }
  // one wave per t; lane reads 8 contiguous bf16 (one b128), reduce via shuffles
  for (int t = wv; t < 1024; t += 8) {
    u32x4 r = *(const u32x4*)(att_h + (size_t)(b * 1024 + t) * 256 + lane * 8);
    float s = 0.f;
#pragma unroll
    for (int j = 0; j < 4; ++j) {
      s += wvv[2 * j]     * tanhf(st[2 * j]     + bf2f(r[j] & 0xffffu));
      s += wvv[2 * j + 1] * tanhf(st[2 * j + 1] + bf2f(r[j] >> 16));
    }
    s = wsum(s);
    if (lane == 0) sc[t] = s + (x_mask[b * 1024 + t] - 1.0f) * 1e30f;
  }
  __syncthreads();
  // softmax: max
  float m = -3.0e38f;
  for (int t = tid; t < 1024; t += 256) m = fmaxf(m, sc[t]);
#pragma unroll
  for (int o = 16; o > 0; o >>= 1) m = fmaxf(m, __shfl_down(m, o, 32));
  if (lane == 0) red[wv] = m;
  __syncthreads();
  float gmax = red[0];
#pragma unroll
  for (int i = 1; i < 8; ++i) gmax = fmaxf(gmax, red[i]);
  __syncthreads();
  // exp + sum
  float s = 0.f;
  for (int t = tid; t < 1024; t += 256) {
    float e = __expf(sc[t] - gmax);
    sc[t] = e;
    s += e;
  }
  s = wsum(s);
  if (lane == 0) red[wv] = s;
  __syncthreads();
  float gsum = red[0];
#pragma unroll
  for (int i = 1; i < 8; ++i) gsum += red[i];
  float inv = 1.0f / gsum;
  // ctx[d] = sum_t alpha_t * eout[b,t,d]
  {
    int d = tid;
    float acc = 0.f;
    for (int t = 0; t < 1024; ++t)
      acc += sc[t] * bf2f((unsigned int)eout_b[(size_t)(b * 1024 + t) * 256 + d]);
    acc *= inv;
    ctxbuf[b * 256 + d] = acc;                                  // unmasked carry
    float ym = y_mask[b * 100 + u + 1];
    dout_b[(size_t)(b * 99 + u) * 768 + 256 + d] = f2bf(acc * ym);
  }
}

// ---------------------------------------------------------------------------
// Decoder LSTM: persistent, 4 WGs, internal loop over u (batches independent).
// TDM variant: per-step A tile (16 rows x 512 halfs, row stride 99*768)
// double-buffered via tensor_load_to_lds; dynamic LDS keeps xA bufs disjoint
// from the gates scratch so DMA overlaps compute.  NOTE: LDS-derived pointers
// are computed inline (never stored in aggregates) so no addrspacecast lands
// in a static initializer.
// ---------------------------------------------------------------------------
__global__ __launch_bounds__(256) void k_dec(
    const unsigned short* WihP, const unsigned short* WhhP,
    const float* bih, const float* bhh, const float* y_mask,
    unsigned short* dout_b) {
#if HAVE_TDM
  extern __shared__ __align__(16) unsigned char dsm[];
  unsigned short* hA = (unsigned short*)(dsm + 32768);        // 16x256 bf16
  float* gts = (float*)(dsm + 40960);                         // 16x1024 f32
#else
  __shared__ __align__(16) unsigned char smem[65536];
  unsigned short* hA = (unsigned short*)(smem + 16384);       // 16x256 bf16
  float* gts = (float*)smem;                                  // reuse
#endif
  const int b0 = blockIdx.x * 16;
  const int tid = threadIdx.x, lane = tid & 31, wv = tid >> 5;
  const v8f z = {0, 0, 0, 0, 0, 0, 0, 0};
  float hreg[16], creg[16];
#pragma unroll
  for (int m = 0; m < 16; ++m) { hreg[m] = 0.f; creg[m] = 0.f; }

#if HAVE_TDM
  const unsigned long long rstride = 99ull * 768ull;          // halfs between rows
  if (wv == 0)
    tdm_load_2d(lds_off_of(dsm), dout_b + (size_t)(b0 * 99) * 768,
                512, 16, rstride);
#endif

  for (int u = 0; u < 99; ++u) {
#if HAVE_TDM
    const unsigned short* xA =
        (const unsigned short*)(dsm + ((unsigned)(u & 1) << 14));  // 16KB bufs
    if (wv == 0) {
      if (u + 1 < 99) {
        tdm_load_2d(lds_off_of(dsm) + ((unsigned)((u + 1) & 1) << 14),
                    dout_b + (size_t)(b0 * 99 + u + 1) * 768, 512, 16, rstride);
        __builtin_amdgcn_s_wait_tensorcnt(1);   // oldest (current tile) done
      } else {
        __builtin_amdgcn_s_wait_tensorcnt(0);
      }
    }
#else
    unsigned short* xA = (unsigned short*)smem;               // 16x512 bf16
    for (int idx = tid; idx < 16 * 512; idx += 256) {
      int m = idx >> 9, k = idx & 511;
      xA[idx] = dout_b[(size_t)((b0 + m) * 99 + u) * 768 + k];   // att_fea
    }
#endif
#pragma unroll
    for (int m = 0; m < 16; ++m) hA[m * 256 + tid] = f2bf(hreg[m]);
    __syncthreads();

    v8f acc[8];
#pragma unroll
    for (int nf = 0; nf < 8; ++nf) acc[nf] = z;
    for (int kt = 0; kt < 16; ++kt) {
      v16bf a = frag_a_rowmaj(xA, 512, 0, kt * 32, lane);
#pragma unroll
      for (int nf = 0; nf < 8; ++nf)
        acc[nf] = wmma(a, frag_b_pack(WihP, 64, kt, wv * 8 + nf, lane), acc[nf]);
    }
    for (int kt = 0; kt < 8; ++kt) {
      v16bf a = frag_a_rowmaj(hA, 256, 0, kt * 32, lane);
#pragma unroll
      for (int nf = 0; nf < 8; ++nf)
        acc[nf] = wmma(a, frag_b_pack(WhhP, 64, kt, wv * 8 + nf, lane), acc[nf]);
    }
    __syncthreads();
    const int n0w = wv * 128;
#pragma unroll
    for (int nf = 0; nf < 8; ++nf) {
      int n = n0w + nf * 16 + (lane & 15);
      float bias = bih[n] + bhh[n];
      int mrow = (lane & 16) ? 8 : 0;
#pragma unroll
      for (int r = 0; r < 8; ++r) gts[(mrow + r) * 1024 + n] = acc[nf][r] + bias;
    }
    __syncthreads();
    {
      int d = tid;
#pragma unroll
      for (int m = 0; m < 16; ++m) {
        int b = b0 + m;
        float gi = gts[m * 1024 + d];
        float gf = gts[m * 1024 + 256 + d];
        float gg = gts[m * 1024 + 512 + d];
        float go = gts[m * 1024 + 768 + d];
        float c = sigm(gf) * creg[m] + sigm(gi) * tanhf(gg);
        float h = sigm(go) * tanhf(c);
        creg[m] = c;
        hreg[m] = h;
        float ym = y_mask[b * 100 + u + 1];
        dout_b[(size_t)(b * 99 + u) * 768 + 512 + d] = f2bf(h * ym);  // dh * ym
      }
    }
    __syncthreads();
  }
}

// ---------------------------------------------------------------------------
// Classifier: logit = dout @ cls_w^T + cls_b  (M=6336, N=4233->4352, K=768)
// A tile via TDM double-buffer when available; B direct from packed fragments.
// ---------------------------------------------------------------------------
__global__ __launch_bounds__(256) void k_cls(const unsigned short* dout_b,
                                             const unsigned short* clsP,
                                             const float* cls_b, float* out) {
  const int m0 = blockIdx.x * 64, n0 = blockIdx.y * 128;
  const int tid = threadIdx.x, lane = tid & 31, wv = tid >> 5;
  const int wm = wv & 1, wn = wv >> 1;               // 2x4 wave grid, 32x32 each
  const v8f z = {0, 0, 0, 0, 0, 0, 0, 0};
  v8f acc[2][2] = {{z, z}, {z, z}};
  const int ntb = blockIdx.y * 8 + wn * 2;

#if HAVE_TDM
  __shared__ __align__(16) unsigned short lA[2][64 * 32];
  if (wv == 0)
    tdm_load_2d(lds_off_of(&lA[0][0]), dout_b + (size_t)m0 * 768, 32, 64, 768);
  for (int kt = 0; kt < 24; ++kt) {
    if (wv == 0) {
      if (kt + 1 < 24) {
        tdm_load_2d(lds_off_of(&lA[(kt + 1) & 1][0]),
                    dout_b + (size_t)m0 * 768 + (kt + 1) * 32, 32, 64, 768);
        __builtin_amdgcn_s_wait_tensorcnt(1);   // oldest (current tile) done
      } else {
        __builtin_amdgcn_s_wait_tensorcnt(0);
      }
    }
    __syncthreads();
    const unsigned short* curA = lA[kt & 1];
#else
  __shared__ __align__(16) unsigned short lA[64 * 32];
  for (int kt = 0; kt < 24; ++kt) {
    int k0 = kt * 32;
    {   // manual A tile (64x32 bf16): b128 copies
      int row = tid >> 2, cg = (tid & 3) * 8;
      const unsigned short* src = dout_b + (size_t)(m0 + row) * 768 + k0 + cg;
      if (kt < 23) __builtin_prefetch(src + 32, 0, 1);
      *(u32x4*)(lA + row * 32 + cg) = *(const u32x4*)src;
    }
    __syncthreads();
    const unsigned short* curA = lA;
#endif
    if (kt < 23)      // prefetch next k-tile of packed B fragments
      __builtin_prefetch(clsP + (((size_t)((kt + 1) * 272 + ntb) * 32 + lane) << 4),
                         0, 1);
    v16bf a0 = frag_a_rowmaj(curA, 32, wm * 32, 0, lane);
    v16bf a1 = frag_a_rowmaj(curA, 32, wm * 32 + 16, 0, lane);
    v16bf b0 = frag_b_pack(clsP, 272, kt, ntb, lane);
    v16bf b1 = frag_b_pack(clsP, 272, kt, ntb + 1, lane);
    acc[0][0] = wmma(a0, b0, acc[0][0]);
    acc[0][1] = wmma(a0, b1, acc[0][1]);
    acc[1][0] = wmma(a1, b0, acc[1][0]);
    acc[1][1] = wmma(a1, b1, acc[1][1]);
    __syncthreads();
  }
#pragma unroll
  for (int fm = 0; fm < 2; ++fm)
#pragma unroll
    for (int fn = 0; fn < 2; ++fn) {
      int n = n0 + wn * 32 + fn * 16 + (lane & 15);
      if (n < 4233) {
        float bias = cls_b[n];
        int mb = m0 + wm * 32 + fm * 16 + ((lane & 16) ? 8 : 0);
#pragma unroll
        for (int r = 0; r < 8; ++r)
          out[(size_t)(mb + r) * 4233 + n] = acc[fm][fn][r] + bias;
      }
    }
}

// ---------------------------------------------------------------------------
extern "C" void kernel_launch(void* const* d_in, const int* in_sizes, int n_in,
                              void* d_out, int out_size, void* d_ws, size_t ws_size,
                              hipStream_t stream) {
  (void)in_sizes; (void)n_in; (void)out_size; (void)ws_size;
  const float* eout    = (const float*)d_in[0];
  const float* x_mask  = (const float*)d_in[1];
  const int*   y       = (const int*)d_in[2];
  const float* y_mask  = (const float*)d_in[3];
  const float* emb     = (const float*)d_in[4];
  const float* w_w     = (const float*)d_in[5];
  const float* w_b     = (const float*)d_in[6];
  const float* v_w     = (const float*)d_in[7];
  const float* v_b     = (const float*)d_in[8];
  const float* w_att_v = (const float*)d_in[9];
  const float* Wih_att = (const float*)d_in[10];
  const float* Whh_att = (const float*)d_in[11];
  const float* bih_att = (const float*)d_in[12];
  const float* bhh_att = (const float*)d_in[13];
  const float* Wih_dec = (const float*)d_in[14];
  const float* Whh_dec = (const float*)d_in[15];
  const float* bih_dec = (const float*)d_in[16];
  const float* bhh_dec = (const float*)d_in[17];
  const float* cls_w   = (const float*)d_in[18];
  const float* cls_b   = (const float*)d_in[19];
  float* out = (float*)d_out;

  char* ws = (char*)d_ws;
  size_t off = 0;
  auto alloc = [&](size_t bytes) -> void* {
    void* p = ws + off;
    off += (bytes + 255) & ~(size_t)255;
    return p;
  };
  unsigned short* att_h   = (unsigned short*)alloc(64ull * 1024 * 256 * 2);
  unsigned short* eout_b  = (unsigned short*)alloc(64ull * 1024 * 256 * 2);
  unsigned short* embed_b = (unsigned short*)alloc(64ull * 99 * 256 * 2);
  unsigned short* WihP_a  = (unsigned short*)alloc(512ull * 1024 * 2);
  unsigned short* WhhP_a  = (unsigned short*)alloc(256ull * 1024 * 2);
  unsigned short* vwP     = (unsigned short*)alloc(256ull * 256 * 2);
  unsigned short* wwP     = (unsigned short*)alloc(256ull * 256 * 2);
  unsigned short* WihP_d  = (unsigned short*)alloc(512ull * 1024 * 2);
  unsigned short* WhhP_d  = (unsigned short*)alloc(256ull * 1024 * 2);
  unsigned short* clsP    = (unsigned short*)alloc(768ull * 4352 * 2);
  unsigned short* dout_b  = (unsigned short*)alloc(6336ull * 768 * 2);
  float* carry = (float*)alloc(4ull * 16384 * 4);   // h | c | ctx | state
  float* hbuf = carry, * cbuf = carry + 16384;
  float* ctxbuf = carry + 2 * 16384, * statebuf = carry + 3 * 16384;

  auto g = [](int n) { return (n + 255) / 256; };
  // weight packs into WMMA-fragment order (one-time prep)
  k_pack<<<g(256 * 256), 256, 0, stream>>>(w_w, wwP, 256, 256, 16);
  k_pack<<<g(256 * 256), 256, 0, stream>>>(v_w, vwP, 256, 256, 16);
  k_pack<<<g(512 * 1024), 256, 0, stream>>>(Wih_att, WihP_a, 1024, 512, 64);
  k_pack<<<g(256 * 1024), 256, 0, stream>>>(Whh_att, WhhP_a, 1024, 256, 64);
  k_pack<<<g(512 * 1024), 256, 0, stream>>>(Wih_dec, WihP_d, 1024, 512, 64);
  k_pack<<<g(256 * 1024), 256, 0, stream>>>(Whh_dec, WhhP_d, 1024, 256, 64);
  k_pack<<<g(768 * 4352), 256, 0, stream>>>(cls_w, clsP, 4233, 768, 272);
  k_cvt<<<g(64 * 1024 * 32), 256, 0, stream>>>(eout, eout_b, 64 * 1024 * 32);
  k_embed<<<g(64 * 99 * 128), 256, 0, stream>>>(y, emb, embed_b);
  k_zero<<<g(4 * 16384), 256, 0, stream>>>(carry, 4 * 16384);

  // att_h GEMM
  k_atth<<<dim3(512, 4), 256, 0, stream>>>(eout, wwP, w_b, att_h);

  // 99-step attention recurrence
  for (int u = 0; u < 99; ++u) {
    k_att_gates<<<4, 256, 0, stream>>>(u, embed_b, WihP_a, WhhP_a, vwP,
                                       bih_att, bhh_att, v_b, y_mask,
                                       hbuf, cbuf, ctxbuf, statebuf, dout_b);
    k_att_score<<<64, 256, 0, stream>>>(u, att_h, eout_b, statebuf, w_att_v,
                                        x_mask, y_mask, ctxbuf, dout_b);
  }

  // decoder LSTM (persistent, TDM double-buffered input tiles) + classifier
  k_dec<<<4, 256, 106496, stream>>>(WihP_d, WhhP_d, bih_dec, bhh_dec, y_mask,
                                    dout_b);
  k_cls<<<dim3(99, 34), 256, 0, stream>>>(dout_b, clsP, cls_b, out);
}
